// Model_39951785787745
// MI455X (gfx1250) — matrix operand order repaired and checked by reading, not compiled, not run on hardware
//
#include <hip/hip_runtime.h>
#include <hip/hip_bf16.h>

typedef __attribute__((ext_vector_type(16))) _Float16 v16h;
typedef __attribute__((ext_vector_type(8)))  float    v8f;

#define T_STEPS 1000
#define BATCH   128
#define N_IN    85
#define N_RNN   512
#define N_OUT   33
#define KPADX   96                 // x part of K padded 85 -> 96 (3 chunks of 32)
#define KTOT    (KPADX + N_RNN)    // 608 = 19 chunks of 32
#define NKB     (KTOT / 32)        // 19
#define NWG     16                 // recurrence workgroups, 32 cols each
#define ALPHA   0.2f

union FragH { uint4 u[2]; v16h v; };

__device__ __forceinline__ float softplus_f(float x) {
    return fmaxf(x, 0.f) + __logf(1.f + __expf(-fabsf(x)));
}

// CDNA5 async global->LDS copy (ASYNCcnt-tracked, no VGPR round trip)
__device__ __forceinline__ void async_ld_b128(uint32_t lds_off, const void* g) {
    asm volatile("global_load_async_to_lds_b128 %0, %1, off"
                 :: "v"(lds_off), "v"((unsigned long long)(uintptr_t)g)
                 : "memory");
}
__device__ __forceinline__ void wait_asynccnt0() {
    asm volatile("s_wait_asynccnt 0x0" ::: "memory");
}

__device__ __forceinline__ v8f wmma_f16(const FragH& a, const FragH& b, v8f c) {
    return __builtin_amdgcn_wmma_f32_16x16x32_f16(false, a.v, false, b.v,
                                                  (short)0, c, false, false);
}

// ---------------------------------------------------------------- init
__global__ void init_kernel(unsigned* bar, _Float16* hf0, _Float16* hf1, float* hstate) {
    int i = blockIdx.x * blockDim.x + threadIdx.x;
    if (i < 64) bar[i] = 0u;
    if (i < BATCH * N_RNN) {
        hf0[i] = (_Float16)0.f;
        hf1[i] = (_Float16)0.f;
        hstate[i] = 0.f;
    }
}

// ---------------------------------------------------------------- recurrence
// 16 persistent WGs x 256 threads (8 waves). WG w owns output cols [32w,32w+32).
// Each wave: one 16-row tile x two 16-col tiles. B (W^T slice) fragments are
// loop-invariant -> held in VGPRs for all 1000 steps; inner loop is A-ds + WMMA.
__global__ __launch_bounds__(256, 1)
void rnn_kernel(const float* __restrict__ x, const float* __restrict__ W,
                const float* __restrict__ b, const float* __restrict__ noise,
                float* __restrict__ h_out,
                _Float16* __restrict__ hf16_0, _Float16* __restrict__ hf16_1,
                float* __restrict__ hstate, unsigned* __restrict__ bar)
{
    extern __shared__ char smem_raw[];
    _Float16* Ax = (_Float16*)smem_raw;                                 // [128][96]
    _Float16* Ah = (_Float16*)(smem_raw + BATCH * KPADX * 2);           // [128][512]
    _Float16* Wt = (_Float16*)(smem_raw + BATCH * (KPADX + N_RNN) * 2); // [32][608]

    const int wg   = blockIdx.x;
    const int tid  = threadIdx.x;
    const int lane = tid & 31;        // wave32
    const int wave = tid >> 5;
    const int n0   = wg * 32;

    // Stage + transpose this WG's W slice (f32 -> f16, K-major).
    for (int i = tid; i < 32 * KTOT; i += 256) {
        int nl = i / KTOT, k = i % KTOT;
        float v = 0.f;
        if (k < N_IN)        v = W[(size_t)k * N_RNN + n0 + nl];
        else if (k >= KPADX) v = W[(size_t)(k - KPADX + N_IN) * N_RNN + n0 + nl];
        Wt[nl * KTOT + k] = (_Float16)v;
    }
    __syncthreads();

    const int bcol = lane & 15;
    const float bn0 = b[n0 + bcol];
    const float bn1 = b[n0 + 16 + bcol];
    const int m0   = wave * 16;
    const int mrow = m0 + bcol;                 // A-fragment row for this lane
    const int aoff = (lane < 16) ? 0 : 16;      // A K offsets, contiguous order k = 16h + s
    const int boff = (lane < 16) ? 0 : 16;      // B K offsets, contiguous order k = 16h + s
    const uint32_t ah_base = (uint32_t)(uintptr_t)Ah;

    // Hoist ALL B fragments into registers (19 kb x 2 col tiles x 8 VGPRs).
    FragH fbr[NKB][2];
    #pragma unroll
    for (int kb = 0; kb < NKB; ++kb) {
        const int k = kb * 32;
        fbr[kb][0].u[0] = *(const uint4*)&Wt[bcol * KTOT + k + boff];
        fbr[kb][0].u[1] = *(const uint4*)&Wt[bcol * KTOT + k + boff + 8];
        fbr[kb][1].u[0] = *(const uint4*)&Wt[(16 + bcol) * KTOT + k + boff];
        fbr[kb][1].u[1] = *(const uint4*)&Wt[(16 + bcol) * KTOT + k + boff + 8];
    }

    for (int t = 0; t < T_STEPS; ++t) {
        const _Float16* hin    = (t & 1) ? hf16_1 : hf16_0;
        _Float16*       hout16 = (t & 1) ? hf16_0 : hf16_1;

        // ---- kick off async h_{t-1} f16 -> LDS (8192 x b128, linear copy)
        {
            const char* hsrc = (const char*)hin;
            for (int i = tid; i < (BATCH * N_RNN * 2) / 16; i += 256)
                async_ld_b128(ah_base + (uint32_t)i * 16u, hsrc + (size_t)i * 16);
        }
        // ---- meanwhile convert x_t f32 -> f16 into Ax with VALU
        const float* xt = x + (size_t)t * BATCH * N_IN;
        for (int i = tid; i < BATCH * KPADX; i += 256) {
            int m = i / KPADX, c = i % KPADX;
            Ax[m * KPADX + c] = (_Float16)(c < N_IN ? xt[m * N_IN + c] : 0.f);
        }
        wait_asynccnt0();
        __syncthreads();

        // ---- prefetch next step's x and this WG's noise columns
        if (t + 1 < T_STEPS) {
            const char* xn = (const char*)(x + (size_t)(t + 1) * BATCH * N_IN);
            for (int off = tid * 128; off < BATCH * N_IN * 4; off += 256 * 128)
                __builtin_prefetch(xn + off, 0, 0);
            const float* nzn = noise + (size_t)(t + 1) * BATCH * N_RNN;
            if (tid < BATCH)
                __builtin_prefetch(&nzn[(size_t)tid * N_RNN + n0], 0, 0);
        }

        // ---- 19 K-chunks: 2 A ds_loads + 2 WMMA each (B already resident)
        v8f acc0 = {}, acc1 = {};
        #pragma unroll
        for (int kb = 0; kb < NKB; ++kb) {
            const int k = kb * 32;
            FragH fa;
            if (kb < KPADX / 32) {
                fa.u[0] = *(const uint4*)&Ax[mrow * KPADX + k + aoff];
                fa.u[1] = *(const uint4*)&Ax[mrow * KPADX + k + aoff + 8];
            } else {
                const int kh = k - KPADX;
                fa.u[0] = *(const uint4*)&Ah[mrow * N_RNN + kh + aoff];
                fa.u[1] = *(const uint4*)&Ah[mrow * N_RNN + kh + aoff + 8];
            }
            acc0 = wmma_f16(fa, fbr[kb][0], acc0);
            acc1 = wmma_f16(fa, fbr[kb][1], acc1);
        }

        // ---- epilogue: gate -> leaky softplus update, write f32 + f16 state
        const float* nz   = noise + (size_t)t * BATCH * N_RNN;
        float*       hrow = h_out + (size_t)t * BATCH * N_RNN;
        #pragma unroll
        for (int nt = 0; nt < 2; ++nt) {
            const v8f   acc = nt ? acc1 : acc0;
            const float bn  = nt ? bn1 : bn0;
            const int   n   = n0 + nt * 16 + bcol;
            #pragma unroll
            for (int r = 0; r < 8; ++r) {
                int m = m0 + r + ((lane >> 4) << 3);
                size_t idx = (size_t)m * N_RNN + n;
                float gate = acc[r] + bn + nz[idx];
                float hnew = (1.f - ALPHA) * hstate[idx] + ALPHA * softplus_f(gate);
                hstate[idx] = hnew;
                hrow[idx]   = hnew;
                hout16[idx] = (_Float16)hnew;
            }
        }

        // ---- grid-wide barrier (monotone ticket counter, re-zeroed by init)
        __threadfence();
        __syncthreads();
        if (tid == 0) {
            __hip_atomic_fetch_add(bar, 1u, __ATOMIC_ACQ_REL, __HIP_MEMORY_SCOPE_AGENT);
            const unsigned target = (unsigned)NWG * (unsigned)(t + 1);
            while (__hip_atomic_load(bar, __ATOMIC_ACQUIRE, __HIP_MEMORY_SCOPE_AGENT) < target) {
                __builtin_amdgcn_s_sleep(1);
            }
        }
        __syncthreads();
    }
}

// ---------------------------------------------------------------- output head
// grid = T WGs; WG t computes y[t] = sigmoid(h[t] @ W_out + b_out) via WMMA,
// N padded 33 -> 48 (3 col tiles), h converted f32->f16 on the fly into LDS.
__global__ __launch_bounds__(256, 1)
void head_kernel(const float* __restrict__ h, const float* __restrict__ Wout,
                 const float* __restrict__ bout, float* __restrict__ y)
{
    extern __shared__ char smem_raw[];
    _Float16* Ahd = (_Float16*)smem_raw;                        // [128][512]
    _Float16* Wt  = (_Float16*)(smem_raw + BATCH * N_RNN * 2);  // [48][512]

    const int t = blockIdx.x;
    const int tid = threadIdx.x, lane = tid & 31, wave = tid >> 5;

    for (int i = tid; i < 48 * N_RNN; i += 256) {
        int n = i / N_RNN, k = i % N_RNN;
        Wt[i] = (_Float16)(n < N_OUT ? Wout[(size_t)k * N_OUT + n] : 0.f);
    }
    const float* ht = h + (size_t)t * BATCH * N_RNN;
    for (int i = tid; i < BATCH * N_RNN; i += 256)
        Ahd[i] = (_Float16)ht[i];
    __syncthreads();

    const int bcol = lane & 15;
    const int m0   = wave * 16;
    const int mrow = m0 + bcol;
    const int aoff = (lane < 16) ? 0 : 16;
    const int boff = (lane < 16) ? 0 : 16;

    #pragma unroll
    for (int nt = 0; nt < 3; ++nt) {
        v8f acc = {};
        #pragma unroll
        for (int kb = 0; kb < N_RNN / 32; ++kb) {
            const int k = kb * 32;
            FragH fa, fb;
            fa.u[0] = *(const uint4*)&Ahd[mrow * N_RNN + k + aoff];
            fa.u[1] = *(const uint4*)&Ahd[mrow * N_RNN + k + aoff + 8];
            fb.u[0] = *(const uint4*)&Wt[(nt * 16 + bcol) * N_RNN + k + boff];
            fb.u[1] = *(const uint4*)&Wt[(nt * 16 + bcol) * N_RNN + k + boff + 8];
            acc = wmma_f16(fa, fb, acc);
        }
        const int n = nt * 16 + bcol;
        if (n < N_OUT) {
            const float bo = bout[n];
            #pragma unroll
            for (int r = 0; r < 8; ++r) {
                int m = m0 + r + ((lane >> 4) << 3);
                float v = acc[r] + bo;
                y[(size_t)t * BATCH * N_OUT + (size_t)m * N_OUT + n] =
                    1.f / (1.f + __expf(-v));
            }
        }
    }
}

// ---------------------------------------------------------------- launch
extern "C" void kernel_launch(void* const* d_in, const int* in_sizes, int n_in,
                              void* d_out, int out_size, void* d_ws, size_t ws_size,
                              hipStream_t stream) {
    const float* x     = (const float*)d_in[0];
    const float* W     = (const float*)d_in[1];
    const float* b     = (const float*)d_in[2];
    const float* Wout  = (const float*)d_in[3];
    const float* bout  = (const float*)d_in[4];
    const float* noise = (const float*)d_in[5];

    float* y     = (float*)d_out;                                   // [T,B,N_OUT]
    float* h_out = y + (size_t)T_STEPS * BATCH * N_OUT;             // [T,B,N_RNN]

    char* ws = (char*)d_ws;
    unsigned*  bar    = (unsigned*)ws;                              // 256 B
    _Float16*  hf0    = (_Float16*)(ws + 256);                      // 131072 B
    _Float16*  hf1    = (_Float16*)(ws + 256 + 131072);             // 131072 B
    float*     hstate = (float*)(ws + 256 + 262144);                // 262144 B

    init_kernel<<<(BATCH * N_RNN + 255) / 256, 256, 0, stream>>>(bar, hf0, hf1, hstate);

    const size_t rnn_smem = (size_t)(BATCH * KPADX + BATCH * N_RNN + 32 * KTOT) * 2; // 194560 B
    rnn_kernel<<<NWG, 256, rnn_smem, stream>>>(x, W, b, noise, h_out,
                                               hf0, hf1, hstate, bar);

    const size_t head_smem = (size_t)(BATCH * N_RNN + 48 * N_RNN) * 2;               // 180224 B
    head_kernel<<<T_STEPS, 256, head_smem, stream>>>(h_out, Wout, bout, y);
}
